// CWSModel_79628693668436
// MI455X (gfx1250) — matrix operand order repaired and checked by reading, not compiled, not run on hardware
//
#include <hip/hip_runtime.h>
#include <hip/hip_bf16.h>

// ---------------------------------------------------------------------------
// CDNA5 (gfx1250) BiLSTM CWS model.
//  - bf16 WMMA (v_wmma_f32_16x16x32_bf16) for input projections + recurrence
//  - fused dual-direction persistent scan (gridDim=2, one WGP per direction),
//    cell state c in registers, h double-buffered in LDS (57KB of 320KB WGP LDS)
//  - global_prefetch of next timestep's gate slab inside the scan
// ---------------------------------------------------------------------------

typedef __bf16 bf16;
typedef __attribute__((ext_vector_type(16))) __bf16 bf16x16;
typedef __attribute__((ext_vector_type(8)))  __bf16 bf16x8;
typedef __attribute__((ext_vector_type(8)))  float  f32x8;

// ---- model dimensions -----------------------------------------------------
constexpr int SEQ  = 512;     // S
constexpr int BAT  = 64;      // B
constexpr int EDIM = 100;     // E
constexpr int HDIM = 200;     // HD (per-direction hidden)
constexpr int GP   = 208;     // padded gate width (multiple of 16)
constexpr int NPAD = 4 * GP;  // 832 padded gate dim (52 n-tiles)
constexpr int NTIL = NPAD / 16;        // 52
constexpr int K0P  = 224;     // padded layer-0 input K (2E=200 -> 224)
constexpr int K1P  = 416;     // padded layer-1 input K (400 -> 416)
constexpr int KHP  = 224;     // padded recurrent K (HD=200 -> 224)
constexpr int NSUB = 510;     // N = S-2
constexpr int WW   = 4;       // W
constexpr int LABN = 5;       // LAB
constexpr int FEAT = 500;     // H + ESUB

__device__ __forceinline__ float sigmoidf_(float x) {
  return 1.0f / (1.0f + __expf(-x));
}
__device__ __forceinline__ int imin_(int a, int b) { return a < b ? a : b; }

// ---------------------------------------------------------------------------
// Pack an (800 x Ksrc) f32 weight into WMMA B-fragment lane order, bf16,
// with gate padding 200->208 rows per gate and K padding to Kpad.
// dst index = ((nt*KT + kt)*32 + lane)*16 + e ; value = W[n][k],
//   n = nt*16 + lane%16,  k = kt*32 + 8*(lane>=16) + (e&7) + 16*(e>>3)   (B-layout mapping)
// ---------------------------------------------------------------------------
__global__ void pack_w_kernel(const float* __restrict__ src,
                              bf16* __restrict__ dst,
                              int Ksrc, int Kpad, long total) {
  long idx = (long)blockIdx.x * blockDim.x + threadIdx.x;
  if (idx >= total) return;
  const int KT = Kpad >> 5;
  int e  = (int)(idx & 15);
  int li = (int)((idx >> 4) & 31);
  long rest = idx >> 9;
  int kt = (int)(rest % KT);
  int nt = (int)(rest / KT);
  int n  = nt * 16 + (li & 15);
  int hf = li >> 4;
  int k  = kt * 32 + 8 * hf + (e & 7) + 16 * (e >> 3);
  int q = n / GP, j = n % GP;
  float v = 0.0f;
  if (j < HDIM && k < Ksrc) v = src[(long)(q * HDIM + j) * Ksrc + k];
  dst[idx] = (bf16)v;
}

__global__ void pack_bias_kernel(const float* __restrict__ src,
                                 float* __restrict__ dst) {
  int n = blockIdx.x * blockDim.x + threadIdx.x;
  if (n >= NPAD) return;
  int q = n / GP, j = n % GP;
  dst[n] = (j < HDIM) ? src[q * HDIM + j] : 0.0f;
}

// ---------------------------------------------------------------------------
// Build X0[(t*64+b)*224 + k] = concat(char_emb, bichar_emb, 0-pad) as bf16
// ---------------------------------------------------------------------------
__global__ void embed_kernel(const int* __restrict__ chars,
                             const int* __restrict__ bichars,
                             const float* __restrict__ embC,
                             const float* __restrict__ embB,
                             bf16* __restrict__ X0) {
  long idx = (long)blockIdx.x * blockDim.x + threadIdx.x;
  const long total = (long)SEQ * BAT * K0P;
  if (idx >= total) return;
  int k = (int)(idx % K0P);
  long r = idx / K0P;
  int b = (int)(r % BAT);
  int t = (int)(r / BAT);
  float v = 0.0f;
  if (k < EDIM)           v = embC[(long)chars[b * SEQ + t] * EDIM + k];
  else if (k < 2 * EDIM)  v = embB[(long)bichars[b * SEQ + t] * EDIM + (k - EDIM)];
  X0[idx] = (bf16)v;
}

__global__ void zero_u32_kernel(unsigned* __restrict__ p, long n) {
  long i = (long)blockIdx.x * blockDim.x + threadIdx.x;
  if (i < n) p[i] = 0u;
}

// ---------------------------------------------------------------------------
// Input projection GEMM: G[(t*64+b), n] = X @ Wp^T + bias
//   M = SEQ*BAT (32768), N = NPAD (832), K = Kpad (224 or 416)
// Each wave computes a 32x64 macro-tile: 8 v_wmma_f32_16x16x32_bf16 per
// K-iter on 6 fragment loads (2 A + 4 B). NPAD = 13 * 64 exactly.
// ---------------------------------------------------------------------------
__global__ __launch_bounds__(256) void gemm_in_kernel(
    const bf16* __restrict__ X, int Kpad,
    const bf16* __restrict__ Wp, const float* __restrict__ bias,
    float* __restrict__ G) {
  const int KT = Kpad >> 5;
  const int NMT = NPAD / 64;  // 13
  int wave = blockIdx.x * (blockDim.x >> 5) + (threadIdx.x >> 5);
  int lane = threadIdx.x & 31;
  int m0 = (wave / NMT) * 32;
  int n0 = (wave % NMT) * 64;
  if (m0 >= SEQ * BAT) return;  // wave-uniform
  int hf = lane >> 4, ln = lane & 15;

  f32x8 acc[2][4];
#pragma unroll
  for (int mm = 0; mm < 2; ++mm)
#pragma unroll
    for (int nn = 0; nn < 4; ++nn)
      acc[mm][nn] = (f32x8){0,0,0,0,0,0,0,0};

  const bf16* xrow0 = X + (long)(m0 + ln) * Kpad;
  const bf16* xrow1 = X + (long)(m0 + 16 + ln) * Kpad;
  const int nt0 = n0 >> 4;

  for (int kt = 0; kt < KT; ++kt) {
    int kb = kt * 32 + 8 * hf;
    bf16x8 lo0 = *(const bf16x8*)(xrow0 + kb);
    bf16x8 hi0 = *(const bf16x8*)(xrow0 + kb + 16);
    bf16x8 lo1 = *(const bf16x8*)(xrow1 + kb);
    bf16x8 hi1 = *(const bf16x8*)(xrow1 + kb + 16);
    bf16x16 a0, a1;
#pragma unroll
    for (int i = 0; i < 8; ++i) {
      a0[i] = lo0[i]; a0[8 + i] = hi0[i];
      a1[i] = lo1[i]; a1[8 + i] = hi1[i];
    }
    bf16x16 bf[4];
#pragma unroll
    for (int nn = 0; nn < 4; ++nn)
      bf[nn] = *(const bf16x16*)(Wp + ((((long)(nt0 + nn) * KT) + kt) * 32 + lane) * 16);
#pragma unroll
    for (int nn = 0; nn < 4; ++nn) {
      acc[0][nn] = __builtin_amdgcn_wmma_f32_16x16x32_bf16(false, a0, false, bf[nn],
                                                           (short)0, acc[0][nn], false, false);
      acc[1][nn] = __builtin_amdgcn_wmma_f32_16x16x32_bf16(false, a1, false, bf[nn],
                                                           (short)0, acc[1][nn], false, false);
    }
  }

#pragma unroll
  for (int nn = 0; nn < 4; ++nn) {
    float bia = bias[n0 + nn * 16 + ln];
#pragma unroll
    for (int r = 0; r < 8; ++r) {
      long row0 = m0 + r + 8 * hf;
      G[row0 * NPAD + n0 + nn * 16 + ln]        = acc[0][nn][r] + bia;
      G[(row0 + 16) * NPAD + n0 + nn * 16 + ln] = acc[1][nn][r] + bia;
    }
  }
}

// ---------------------------------------------------------------------------
// Recurrent scan, both directions fused: blockIdx.x (+dsel) picks direction,
// each direction is one persistent 13-wave workgroup on its own WGP.
// Wave w owns column-tile ct=w (16 gate columns) across ALL four gates, so
// the cell state c stays in registers for the whole sequence.
//   per step: g = G[t] + h_{t-1} @ Whh^T   (WMMA, h from LDS, Whh from L2)
// h is double-buffered in LDS: 2 x 64 x 224 bf16 = 57344 B.
// Next step's gate slab (64x832 f32 = 416 thr x 4 x 128B) is prefetched.
// ---------------------------------------------------------------------------
__global__ __launch_bounds__(416) void lstm_scan_kernel(
    const float* __restrict__ G0, const float* __restrict__ G1,
    const bf16* __restrict__ Whp0, const bf16* __restrict__ Whp1,
    bf16* __restrict__ xl_out,                           // (SEQ*BAT,K1P) or null
    float* __restrict__ hf0, float* __restrict__ hf1,    // (SEQ*BAT,HDIM) or null
    int dsel) {
  const int d = (int)blockIdx.x + dsel;          // 0 = fwd, 1 = bwd
  const float* __restrict__ G    = d ? G1 : G0;
  const bf16*  __restrict__ Whhp = d ? Whp1 : Whp0;
  float* __restrict__ hf_out     = d ? hf1 : hf0;
  const int dir  = d ? -1 : +1;
  const int xoff = d ? HDIM : 0;

  __shared__ __attribute__((aligned(16))) bf16 hbuf[2][BAT][KHP];
  for (int i = threadIdx.x; i < 2 * BAT * KHP; i += blockDim.x)
    ((bf16*)hbuf)[i] = (bf16)0.0f;
  __syncthreads();

  const int lane = threadIdx.x & 31;
  const int ct   = threadIdx.x >> 5;   // 0..12
  const int hfh  = lane >> 4, ln = lane & 15;
  const int j    = ct * 16 + ln;       // gate column (0..207)

  float c[4][8];
#pragma unroll
  for (int m = 0; m < 4; ++m)
#pragma unroll
    for (int r = 0; r < 8; ++r) c[m][r] = 0.0f;

  for (int step = 0; step < SEQ; ++step) {
    int t = (dir > 0) ? step : (SEQ - 1 - step);
    int cur = step & 1, nxt = cur ^ 1;

    // prefetch next timestep's gate slab (exactly 64*832 floats)
    {
      int tn = (dir > 0) ? imin_(t + 1, SEQ - 1) : (t > 0 ? t - 1 : 0);
      const float* gp = G + (long)tn * BAT * NPAD + (long)threadIdx.x * 128;
#pragma unroll
      for (int i = 0; i < 4; ++i)
        __builtin_prefetch((const void*)(gp + i * 32), 0, 1);
    }

#pragma unroll 1
    for (int m = 0; m < 4; ++m) {
      int arow = m * 16 + ln;
      bf16x16 a[7];
#pragma unroll
      for (int kk = 0; kk < 7; ++kk) {
        const bf16* p = &hbuf[cur][arow][kk * 32 + 8 * hfh];
        bf16x8 lo = *(const bf16x8*)p;
        bf16x8 hi = *(const bf16x8*)(p + 16);
#pragma unroll
        for (int i = 0; i < 8; ++i) { a[kk][i] = lo[i]; a[kk][8 + i] = hi[i]; }
      }

      float gate[4][8];
#pragma unroll 1
      for (int q = 0; q < 4; ++q) {
        f32x8 acc = {0,0,0,0,0,0,0,0};
        const bf16* wb = Whhp + (((long)(q * 13 + ct) * 7) * 32 + lane) * 16;
#pragma unroll
        for (int kk = 0; kk < 7; ++kk) {
          bf16x16 bfr = *(const bf16x16*)(wb + (long)kk * 512);
          acc = __builtin_amdgcn_wmma_f32_16x16x32_bf16(false, a[kk], false, bfr,
                                                        (short)0, acc, false, false);
        }
#pragma unroll
        for (int r = 0; r < 8; ++r) {
          int brow = m * 16 + r + 8 * hfh;
          gate[q][r] = acc[r] + G[((long)t * BAT + brow) * NPAD + q * GP + ct * 16 + ln];
        }
      }

#pragma unroll
      for (int r = 0; r < 8; ++r) {
        int brow = m * 16 + r + 8 * hfh;
        float i_ = sigmoidf_(gate[0][r]);
        float f_ = sigmoidf_(gate[1][r]);
        float g_ = tanhf(gate[2][r]);
        float o_ = sigmoidf_(gate[3][r]);
        float cc = f_ * c[m][r] + i_ * g_;
        c[m][r] = cc;
        float h = o_ * tanhf(cc);
        if (j >= HDIM) h = 0.0f;  // keep padded columns exactly zero
        hbuf[nxt][brow][j] = (bf16)h;
        if (j < HDIM) {
          long row = (long)t * BAT + brow;
          if (xl_out) xl_out[row * K1P + xoff + j] = (bf16)h;
          if (hf_out) hf_out[row * HDIM + j] = h;
        }
      }
    }
    __syncthreads();
  }
}

// ---------------------------------------------------------------------------
// Span features + FFN head: out[b,t,w,:] = feats @ ffn_W^T + ffn_b
// ---------------------------------------------------------------------------
__global__ __launch_bounds__(256) void span_ffn_kernel(
    const float* __restrict__ HF, const float* __restrict__ HB,
    const int* __restrict__ subw, const float* __restrict__ embS,
    const float* __restrict__ ffnW, const float* __restrict__ ffnb,
    float* __restrict__ out) {
  __shared__ float Wl[LABN * FEAT];
  __shared__ float bl[LABN];
  for (int i = threadIdx.x; i < LABN * FEAT; i += blockDim.x) Wl[i] = ffnW[i];
  if (threadIdx.x < LABN) bl[threadIdx.x] = ffnb[threadIdx.x];
  __syncthreads();

  long p = (long)blockIdx.x * blockDim.x + threadIdx.x;
  const long TOT = (long)BAT * NSUB * WW;
  if (p >= TOT) return;
  int w = (int)(p % WW);
  long r = p / WW;
  int t = (int)(r % NSUB);
  int b = (int)(r / NSUB);

  float acc[LABN];
#pragma unroll
  for (int l = 0; l < LABN; ++l) acc[l] = bl[l];

  if (t + w < NSUB) {
    int i_f = imin_(t + w + 1, SEQ - 1);
    int i_b = imin_(t + w + 2, SEQ - 1);
    const float* f1 = HF + ((long)i_f * BAT + b) * HDIM;
    const float* f0 = HF + ((long)t   * BAT + b) * HDIM;
    const float* b1 = HB + ((long)(t + 1) * BAT + b) * HDIM;
    const float* b0 = HB + ((long)i_b * BAT + b) * HDIM;
    for (int jj = 0; jj < HDIM; ++jj) {
      float sf = f1[jj] - f0[jj];
      float sb = b1[jj] - b0[jj];
#pragma unroll
      for (int l = 0; l < LABN; ++l) {
        acc[l] += sf * Wl[l * FEAT + jj];
        acc[l] += sb * Wl[l * FEAT + HDIM + jj];
      }
    }
  }
  int sid = subw[((long)b * NSUB + t) * WW + w];
  const float* ev = embS + (long)sid * EDIM;
  for (int jj = 0; jj < EDIM; ++jj) {
    float e = ev[jj];
#pragma unroll
    for (int l = 0; l < LABN; ++l) acc[l] += e * Wl[l * FEAT + 2 * HDIM + jj];
  }
  float* op = out + p * LABN;
#pragma unroll
  for (int l = 0; l < LABN; ++l) op[l] = acc[l];
}

// ---------------------------------------------------------------------------
// Host launcher
// ---------------------------------------------------------------------------
extern "C" void kernel_launch(void* const* d_in, const int* in_sizes, int n_in,
                              void* d_out, int out_size, void* d_ws, size_t ws_size,
                              hipStream_t stream) {
  (void)out_size;
  if (n_in < 20) return;

  // ---- inputs (dict order; nested params assumed jax-sorted, detected) ----
  const int* chars   = (const int*)d_in[0];
  const int* bichars = (const int*)d_in[1];
  const int* subw    = (const int*)d_in[2];
  const float *embC, *embB;
  if (in_sizes[3] == 10000 * 100) { embC = (const float*)d_in[3]; embB = (const float*)d_in[4]; }
  else                            { embB = (const float*)d_in[3]; embC = (const float*)d_in[4]; }
  const float* embS = (const float*)d_in[5];
  const float* ffnW = (const float*)d_in[6];
  const float* ffnb = (const float*)d_in[7];

  // lstm params at indices 8..19; detect key ordering from layer-1 Wih size
  const float *Wih[2][2], *Whh[2][2], *bb[2][2];  // [layer][0=fwd,1=bwd]
  bool insertion = (in_sizes[14] == 800 * 400);
  for (int l = 0; l < 2; ++l) {
    int base = 8 + l * 6;
    if (insertion) {  // fwd{Wih,Whh,b}, bwd{Wih,Whh,b}
      Wih[l][0] = (const float*)d_in[base + 0];
      Whh[l][0] = (const float*)d_in[base + 1];
      bb [l][0] = (const float*)d_in[base + 2];
      Wih[l][1] = (const float*)d_in[base + 3];
      Whh[l][1] = (const float*)d_in[base + 4];
      bb [l][1] = (const float*)d_in[base + 5];
    } else {          // sorted: bwd{Whh,Wih,b}, fwd{Whh,Wih,b}
      Whh[l][1] = (const float*)d_in[base + 0];
      Wih[l][1] = (const float*)d_in[base + 1];
      bb [l][1] = (const float*)d_in[base + 2];
      Whh[l][0] = (const float*)d_in[base + 3];
      Wih[l][0] = (const float*)d_in[base + 4];
      bb [l][0] = (const float*)d_in[base + 5];
    }
  }

  // ---- workspace carve-up -------------------------------------------------
  char* ws = (char*)d_ws;
  size_t off = 0;
  auto alloc = [&](size_t bytes) {
    size_t o = off;
    off = (off + bytes + 255) & ~(size_t)255;
    return o;
  };
  const size_t Gbytes = (size_t)SEQ * BAT * NPAD * 4;
  bf16*  X0 = (bf16*)(ws + alloc((size_t)SEQ * BAT * K0P * 2));
  bf16*  XL = (bf16*)(ws + alloc((size_t)SEQ * BAT * K1P * 2));
  float* Gf = (float*)(ws + alloc(Gbytes));
  float* HF = (float*)(ws + alloc((size_t)SEQ * BAT * HDIM * 4));
  float* HB = (float*)(ws + alloc((size_t)SEQ * BAT * HDIM * 4));
  bf16*  Wp_ih[2][2];
  bf16*  Wp_hh[2][2];
  float* bp[2][2];
  for (int l = 0; l < 2; ++l)
    for (int d = 0; d < 2; ++d) {
      int Kpad = (l == 0) ? K0P : K1P;
      Wp_ih[l][d] = (bf16*)(ws + alloc((size_t)NPAD * Kpad * 2));
      Wp_hh[l][d] = (bf16*)(ws + alloc((size_t)NPAD * KHP * 2));
      bp[l][d]    = (float*)(ws + alloc((size_t)NPAD * 4));
    }
  // second gate buffer -> both directions run concurrently (2 workgroups)
  bool parallel_dirs = (ws_size >= off + Gbytes + 256);
  float* Gb = parallel_dirs ? (float*)(ws + alloc(Gbytes)) : Gf;

  // ---- weight packing -----------------------------------------------------
  for (int l = 0; l < 2; ++l)
    for (int d = 0; d < 2; ++d) {
      int Ksrc = (l == 0) ? 200 : 400;
      int Kpad = (l == 0) ? K0P : K1P;
      long tot_ih = (long)NTIL * (Kpad >> 5) * 512;
      long tot_hh = (long)NTIL * (KHP  >> 5) * 512;
      pack_w_kernel<<<(int)((tot_ih + 255) / 256), 256, 0, stream>>>(
          Wih[l][d], Wp_ih[l][d], Ksrc, Kpad, tot_ih);
      pack_w_kernel<<<(int)((tot_hh + 255) / 256), 256, 0, stream>>>(
          Whh[l][d], Wp_hh[l][d], HDIM, KHP, tot_hh);
      pack_bias_kernel<<<(NPAD + 255) / 256, 256, 0, stream>>>(bb[l][d], bp[l][d]);
    }

  // ---- embeddings + clear layer-1 input buffer ---------------------------
  {
    long tot = (long)SEQ * BAT * K0P;
    embed_kernel<<<(int)((tot + 255) / 256), 256, 0, stream>>>(
        chars, bichars, embC, embB, X0);
    long nz = (long)SEQ * BAT * K1P * 2 / 4;
    zero_u32_kernel<<<(int)((nz + 255) / 256), 256, 0, stream>>>((unsigned*)XL, nz);
  }

  // ---- BiLSTM layers ------------------------------------------------------
  const int gemm_blocks = (SEQ * BAT / 32) * (NPAD / 64) / 8;  // 1664
  for (int l = 0; l < 2; ++l) {
    const bf16* Xin = (l == 0) ? X0 : XL;
    int Kpad        = (l == 0) ? K0P : K1P;
    bf16*  xl = (l == 0) ? XL : nullptr;
    float* h0 = (l == 0) ? nullptr : HF;
    float* h1 = (l == 0) ? nullptr : HB;
    if (parallel_dirs) {
      gemm_in_kernel<<<gemm_blocks, 256, 0, stream>>>(Xin, Kpad, Wp_ih[l][0], bp[l][0], Gf);
      gemm_in_kernel<<<gemm_blocks, 256, 0, stream>>>(Xin, Kpad, Wp_ih[l][1], bp[l][1], Gb);
      lstm_scan_kernel<<<2, 416, 0, stream>>>(Gf, Gb, Wp_hh[l][0], Wp_hh[l][1],
                                              xl, h0, h1, 0);
    } else {  // single gate buffer: serialize directions
      gemm_in_kernel<<<gemm_blocks, 256, 0, stream>>>(Xin, Kpad, Wp_ih[l][0], bp[l][0], Gf);
      lstm_scan_kernel<<<1, 416, 0, stream>>>(Gf, Gf, Wp_hh[l][0], Wp_hh[l][1],
                                              xl, h0, h1, 0);
      gemm_in_kernel<<<gemm_blocks, 256, 0, stream>>>(Xin, Kpad, Wp_ih[l][1], bp[l][1], Gb);
      lstm_scan_kernel<<<1, 416, 0, stream>>>(Gf, Gf, Wp_hh[l][0], Wp_hh[l][1],
                                              xl, h0, h1, 1);
    }
  }

  // ---- span features + FFN head ------------------------------------------
  {
    long tot = (long)BAT * NSUB * WW;
    span_ffn_kernel<<<(int)((tot + 255) / 256), 256, 0, stream>>>(
        HF, HB, subw, embS, ffnW, ffnb, (float*)d_out);
  }
}
